// criss_ctrgcn_31860067402319
// MI455X (gfx1250) — matrix operand order repaired and checked by reading, not compiled, not run on hardware
//
#include <hip/hip_runtime.h>
#include <hip/hip_bf16.h>

typedef __bf16 bf16;
typedef __attribute__((ext_vector_type(16))) __bf16 v16bf;
typedef __attribute__((ext_vector_type(8)))  __bf16 v8bf;
typedef __attribute__((ext_vector_type(8)))  float  v8f;
typedef __attribute__((ext_vector_type(4)))  unsigned int u32x4;
typedef __attribute__((ext_vector_type(8)))  unsigned int u32x8;

#define EPSV 1e-5f
#define NN 64
#define CC 256
#define TT 128
#define VV 25
#define VP 32
#define K3 3
#define KM 192
#define MID 64
#define MIP 24
#define TV (TT*VV)   // 3200

// ---- WMMA operand loaders (contiguous k = 16h + s, same order in A and B) ----
// A 16x32 bf16, row-major MxK: lane m=lid%16; lanes<16 -> K0..15, lanes>=16 -> K16..31
__device__ inline v16bf load_a16x32(const bf16* base, int ld) {
  int lane = threadIdx.x & 31;
  const bf16* rp = base + (lane & 15) * ld;
  int kb = (lane < 16) ? 0 : 16;
  return *(const v16bf*)(rp + kb);
}
// B 32x16 bf16 stored transposed [N][K] (K contiguous): lanes<16 -> K0..15, lanes>=16 -> K16..31
__device__ inline v16bf load_b32x16(const bf16* baseT, int ld) {
  int lane = threadIdx.x & 31;
  return *(const v16bf*)(baseT + (lane & 15) * ld + ((lane < 16) ? 0 : 16));
}
__device__ inline v8f wmma_bf16(v16bf a, v16bf b, v8f c) {
  return __builtin_amdgcn_wmma_f32_16x16x32_bf16(false, a, false, b, (short)0, c, false, false);
}

// ---- Tensor Data Mover: 2D tile Global -> LDS (ISA 08_async_tensor D# layout) ----
// Generic LDS pointer low 32 bits == LDS byte offset (flat aperture truncation).
__device__ inline unsigned lds_addr_of(const void* p) {
  return (unsigned)(unsigned long long)p;
}
// tile_d0_u8 / stride_u8 in 8-byte units (data_size code 3). Groups 2/3 zeroed (<=2D tile).
__device__ inline void tdm_load_2d(unsigned lds_off, const void* gptr,
                                   unsigned tile_d0_u8, unsigned tile_d1,
                                   unsigned stride_u8) {
  unsigned long long ga = (unsigned long long)gptr;
  u32x4 g0;
  g0[0] = 1u;                                             // count=1, user descriptor
  g0[1] = lds_off;                                        // lds_addr (bytes)
  g0[2] = (unsigned)ga;                                   // global_addr[31:0]
  g0[3] = (unsigned)((ga >> 32) & 0x01FFFFFFu) | (2u << 30); // addr[56:32] | type=2
  u32x8 g1;
  g1[0] = 3u << 16;                                       // workgroup_mask=0, data_size=3 (8B)
  g1[1] = (tile_d0_u8 & 0xFFFFu) << 16;                   // tensor_dim0[15:0] (== tile_d0)
  g1[2] = ((tile_d0_u8 >> 16) & 0xFFFFu) | ((tile_d1 & 0xFFFFu) << 16); // dim0 hi | tensor_dim1 lo
  g1[3] = ((tile_d1 >> 16) & 0xFFFFu) | ((tile_d0_u8 & 0xFFFFu) << 16); // dim1 hi | tile_dim0
  g1[4] = (tile_d1 & 0xFFFFu);                            // tile_dim1 | tile_dim2=0
  g1[5] = stride_u8;                                      // tensor_dim0_stride[31:0]
  g1[6] = 0u;                                             // stride hi | dim1_stride lo
  g1[7] = 0u;                                             // dim1_stride hi
  u32x4 gz;
  gz[0] = 0u; gz[1] = 0u; gz[2] = 0u; gz[3] = 0u;
  asm volatile("tensor_load_to_lds %0, %1, %2, %3"
               :: "s"(g0), "s"(g1), "s"(gz), "s"(gz)
               : "memory");
}

// ---- prep: fold BN consts, reduce conv3 weights, convert GEMM weights to bf16 ----
__global__ void k_prep(const float* pre_w, const float* pre_b, const float* bn_pre,
                       const float* se1_b, const float* se_bn1,
                       const float* c3_w, const float* c3_b,
                       const float* post_w, const float* post_b, const float* bn_out,
                       float* pre_sc, float* pre_sh, float* out_sc, float* out_sh,
                       float* se1_sc, float* se1_sh, float* w3r, float* b3r,
                       bf16* wpre, bf16* wpost) {
  int tid = threadIdx.x;
  for (int o = tid; o < KM; o += 256) {
    float s = bn_pre[0*KM+o] * rsqrtf(bn_pre[3*KM+o] + EPSV);
    pre_sc[o] = s;
    pre_sh[o] = (pre_b[o] - bn_pre[2*KM+o]) * s + bn_pre[1*KM+o];
  }
  for (int o = tid; o < CC; o += 256) {
    float s = bn_out[0*CC+o] * rsqrtf(bn_out[3*CC+o] + EPSV);
    out_sc[o] = s;
    out_sh[o] = (post_b[o] - bn_out[2*CC+o]) * s + bn_out[1*CC+o];
  }
  for (int o = tid; o < MIP; o += 256) {
    float s = se_bn1[0*MIP+o] * rsqrtf(se_bn1[3*MIP+o] + EPSV);
    se1_sc[o] = s;
    se1_sh[o] = (se1_b[o] - se_bn1[2*MIP+o]) * s + se_bn1[1*MIP+o];
  }
  for (int i = tid; i < K3*CC; i += 256) {       // conv3 folded with channel-mean
    int k = i / CC, c = i % CC;
    float a = 0.f;
    for (int m = 0; m < MID; ++m) a += c3_w[(k*MID+m)*CC + c];
    w3r[i] = a / MID;
  }
  if (tid < K3) {
    float a = 0.f;
    for (int m = 0; m < MID; ++m) a += c3_b[tid*MID + m];
    b3r[tid] = a / MID;
  }
  for (int i = tid; i < KM*CC; i += 256) wpre[i]  = (bf16)pre_w[i];
  for (int i = tid; i < CC*KM; i += 256) wpost[i] = (bf16)post_w[i];
}

// ---- x (n,c,tv) f32 -> xb (n,tv,c) bf16 ----
__global__ void k_cvt(const float* x, bf16* xb) {
  int n = blockIdx.y;
  int j = blockIdx.x * blockDim.x + threadIdx.x;
  if (j >= TV) return;
  const float* src = x + (size_t)n * CC * TV + j;
  bf16* dst = xb + ((size_t)n * TV + j) * CC;
  for (int c = 0; c < CC; ++c) dst[c] = (bf16)src[(size_t)c * TV];
}

// ---- temporal gate softmax + tmp[n,c,v] ----
__global__ void k_gate_tmp(const float* x, float* tmp) {
  __shared__ float red[256];
  __shared__ float g[TT];
  int n = blockIdx.x, tid = threadIdx.x;
  const float* xn = x + (size_t)n * CC * TV;
  for (int t = 0; t < TT; ++t) {
    const float* row = xn + (size_t)tid * TV + t * VV;
    float p = 0.f;
    for (int v = 0; v < VV; ++v) p += row[v];
    red[tid] = p;
    __syncthreads();
    for (int s = 128; s > 0; s >>= 1) {
      if (tid < s) red[tid] += red[tid + s];
      __syncthreads();
    }
    if (tid == 0) g[t] = red[0] / (float)(CC * VV);
    __syncthreads();
  }
  if (tid == 0) {
    float mx = g[0];
    for (int t = 1; t < TT; ++t) mx = fmaxf(mx, g[t]);
    float sum = 0.f;
    for (int t = 0; t < TT; ++t) { g[t] = expf(g[t] - mx); sum += g[t]; }
    float inv = 1.f / sum;
    for (int t = 0; t < TT; ++t) g[t] *= inv;
  }
  __syncthreads();
  int c = tid;
  float acc[VV];
  for (int v = 0; v < VV; ++v) acc[v] = 0.f;
  for (int t = 0; t < TT; ++t) {
    float gt = g[t];
    const float* row = xn + (size_t)c * TV + t * VV;
    for (int v = 0; v < VV; ++v) acc[v] += gt * row[v];
  }
  for (int v = 0; v < VV; ++v) tmp[((size_t)n * CC + c) * VV + v] = acc[v] / (float)TT;
}

// ---- x3[n,k,t,v] via reduced conv3 ----
__global__ void k_x3(const float* x, const float* w3r, const float* b3r, float* x3) {
  int n = blockIdx.y, t = blockIdx.x, tid = threadIdx.x;
  if (tid >= K3 * VV) return;
  int k = tid / VV, v = tid % VV;
  const float* xn = x + (size_t)n * CC * TV + t * VV + v;
  const float* w = w3r + k * CC;
  float a = b3r[k];
  for (int c = 0; c < CC; ++c) a += w[c] * xn[(size_t)c * TV];
  x3[(((size_t)n * K3 + k) * TT + t) * VV + v] = a;
}

// ---- AdynT[n,kc][w][v] bf16, zero-padded to 32x32 ----
__global__ void k_adyn(const float* tmp, const float* c1_w, const float* c1_b,
                       const float* c2_w, const float* c2_b,
                       const float* A, const float* alpha, bf16* AdynT) {
  __shared__ float x1[KM * VV];
  __shared__ float x2[KM * VV];
  int n = blockIdx.x, tid = threadIdx.x;
  const float* tn = tmp + (size_t)n * CC * VV;
  for (int i = tid; i < KM * VV; i += 256) {
    int o = i / VV, v = i % VV;
    float a1 = c1_b[o], a2 = c2_b[o];
    const float* w1 = c1_w + o * CC;
    const float* w2 = c2_w + o * CC;
    for (int c = 0; c < CC; ++c) {
      float xv = tn[c * VV + v];
      a1 += w1[c] * xv;
      a2 += w2[c] * xv;
    }
    x1[i] = a1; x2[i] = a2;
  }
  __syncthreads();
  float al = alpha[0];
  for (int i = tid; i < K3 * MID * VP * VP; i += 256) {
    int kc = i >> 10;
    int r = i & 1023;
    int w = r >> 5, v = r & 31;
    int k = kc / MID;
    float val = 0.f;
    if (v < VV && w < VV)
      val = tanhf(x1[kc * VV + v] - x2[kc * VV + w]) * al + A[(k * VV + v) * VV + w];
    AdynT[((size_t)(n * KM + kc) << 10) + r] = (bf16)val;
  }
}

// ---- A1T[n,k,t][w][v] bf16, zero-padded ----
__global__ void k_a1(const float* x3, bf16* A1T) {
  __shared__ float xr[K3 * VV];
  int n = blockIdx.y, t = blockIdx.x, tid = threadIdx.x;
  if (tid < K3 * VV) {
    int k = tid / VV, v = tid % VV;
    xr[tid] = x3[(((size_t)n * K3 + k) * TT + t) * VV + v];
  }
  __syncthreads();
  for (int i = tid; i < K3 * VP * VP; i += 256) {
    int k = i >> 10, r = i & 1023, w = r >> 5, v = r & 31;
    float val = 0.f;
    if (v < VV && w < VV) val = tanhf(xr[k * VV + v] - xr[k * VV + w]);
    A1T[((size_t)((n * K3 + k) * TT + t) << 10) + r] = (bf16)val;
  }
}

// ---- pre GEMM: pre_s[n,km,t,vp] = relu(bn(pre_w @ x)), bf16 WMMA, TDM-staged B ----
__global__ void k_pre_gemm(const bf16* xb, const bf16* wpre,
                           const float* pre_sc, const float* pre_sh, bf16* pre_s) {
  __shared__ bf16 Bs[16 * CC];                    // 8KB B tile [col][K]
  int n = blockIdx.y, nt = blockIdx.x;            // 200 tiles of 16 columns (tv)
  int wave = threadIdx.x >> 5, lane = threadIdx.x & 31;
  const bf16* bbase = xb + ((size_t)n * TV + nt * 16) * CC;
  if (threadIdx.x < 32) {                         // wave 0 issues the DMA
    tdm_load_2d(lds_addr_of(Bs), bbase, (16 * CC * 2) / 8, 1, (16 * CC * 2) / 8);
    __builtin_amdgcn_s_wait_tensorcnt(0);
  }
  __syncthreads();
  v8f zero = {};
  v8f acc[3];
#pragma unroll
  for (int i = 0; i < 3; ++i) acc[i] = zero;
  for (int kt = 0; kt < CC / 32; ++kt) {
    v16bf b = load_b32x16(Bs + kt * 32, CC);
#pragma unroll
    for (int i = 0; i < 3; ++i) {
      int mt = wave + i * 4;                      // 12 m-tiles over 4 waves
      v16bf a = load_a16x32(wpre + mt * 16 * CC + kt * 32, CC);
      acc[i] = wmma_bf16(a, b, acc[i]);
    }
  }
#pragma unroll
  for (int i = 0; i < 3; ++i) {
    int mt = wave + i * 4;
#pragma unroll
    for (int r = 0; r < 8; ++r) {
      int m = mt * 16 + r + ((lane < 16) ? 0 : 8);
      int j = nt * 16 + (lane & 15);
      int t = j / VV, v = j % VV;
      float val = fmaxf(acc[i][r] * pre_sc[m] + pre_sh[m], 0.f);
      pre_s[(((size_t)n * KM + m) * TT + t) * VP + v] = (bf16)val;
    }
  }
}

// ---- SE pooled means ----
__global__ void k_xhxw(const bf16* pre_s, float* xh, float* xw) {
  __shared__ float tile[TT * VV];
  int n = blockIdx.y, ch = blockIdx.x, tid = threadIdx.x;   // 128 threads, one per t
  const bf16* src = pre_s + ((size_t)n * KM + ch) * TT * VP;
  float rs = 0.f;
  for (int v = 0; v < VV; ++v) {
    float val = (float)src[tid * VP + v];
    tile[tid * VV + v] = val;
    rs += val;
  }
  xh[((size_t)n * KM + ch) * TT + tid] = rs / (float)VV;
  __syncthreads();
  if (tid < VV) {
    float cs = 0.f;
    for (int t = 0; t < TT; ++t) cs += tile[t * VV + tid];
    xw[((size_t)n * KM + ch) * VV + tid] = cs / (float)TT;
  }
}

// ---- SE MLP: y = hswish(bn1(..)), a_h/a_w = sigmoid(..) ----
__global__ void k_se(const float* xh, const float* xw,
                     const float* se1_w, const float* se1_sc, const float* se1_sh,
                     const float* seh_w, const float* seh_b,
                     const float* sew_w, const float* sew_b,
                     float* a_h, float* a_w) {
  __shared__ float y[MIP * 153];
  int n = blockIdx.x, tid = threadIdx.x;
  const float* xhn = xh + (size_t)n * KM * TT;
  const float* xwn = xw + (size_t)n * KM * VV;
  for (int i = tid; i < MIP * 153; i += 256) {
    int p = i / 153, j = i % 153;
    const float* w = se1_w + p * KM;
    float a = 0.f;
    if (j < TT) { for (int ch = 0; ch < KM; ++ch) a += w[ch] * xhn[ch * TT + j]; }
    else { int v = j - TT; for (int ch = 0; ch < KM; ++ch) a += w[ch] * xwn[ch * VV + v]; }
    float t = a * se1_sc[p] + se1_sh[p];
    y[i] = t * fminf(fmaxf(t + 3.f, 0.f), 6.f) * (1.f / 6.f);
  }
  __syncthreads();
  for (int i = tid; i < KM * TT; i += 256) {
    int ch = i / TT, t = i % TT;
    float a = seh_b[ch];
    for (int p = 0; p < MIP; ++p) a += seh_w[ch * MIP + p] * y[p * 153 + t];
    a_h[(size_t)n * KM * TT + i] = 1.f / (1.f + expf(-a));
  }
  for (int i = tid; i < KM * VV; i += 256) {
    int ch = i / VV, v = i % VV;
    float a = sew_b[ch];
    for (int p = 0; p < MIP; ++p) a += sew_w[ch * MIP + p] * y[p * 153 + TT + v];
    a_w[(size_t)n * KM * VV + i] = 1.f / (1.f + expf(-a));
  }
}

// ---- apply SE scale in place; also zero v-padding ----
__global__ void k_scale(bf16* pre_s, const float* a_h, const float* a_w) {
  size_t row = (size_t)blockIdx.x * blockDim.x + threadIdx.x;
  if (row >= (size_t)NN * KM * TT) return;
  int t = (int)(row % TT);
  size_t nch = row / TT;
  int ch = (int)(nch % KM);
  int n = (int)(nch / KM);
  float ah = a_h[((size_t)n * KM + ch) * TT + t];
  const float* aw = a_w + ((size_t)n * KM + ch) * VV;
  bf16* p = pre_s + row * VP;
  for (int v = 0; v < VP; ++v) {
    float val = (v < VV) ? (float)p[v] * ah * aw[v] : 0.f;
    p[v] = (bf16)val;
  }
}

// ---- core einsums: out + pai0*out1, WMMA, TDM-staged pre_s tile ----
__global__ void k_einsum(const bf16* pre_s, const bf16* AdynT, const bf16* A1T,
                         const float* pai, bf16* outc) {
  extern __shared__ char smem[];
  float* lds = (float*)smem;                      // [64 c][16 t][32 w] f32 = 128 KB
  bf16*  Ps  = (bf16*)(smem + 64 * 16 * VP * sizeof(float)); // [64 c][16 t * 32 v] = 64 KB
  int tt = blockIdx.x, k = blockIdx.y, n = blockIdx.z;
  int wave = threadIdx.x >> 5, lane = threadIdx.x & 31;
  int t0 = tt * 16;
  // DMA pre_s tile: 64 rows of 1KB (16t x 32v bf16), row stride TT*VP*2 bytes
  const bf16* gsrc = pre_s + ((size_t)(n * KM + k * MID) * TT + t0) * VP;
  if (threadIdx.x < 32) {
    tdm_load_2d(lds_addr_of(Ps), gsrc, (16 * VP * 2) / 8, MID, (TT * VP * 2) / 8);
    __builtin_amdgcn_s_wait_tensorcnt(0);
  }
  __syncthreads();
  v8f zero = {};
  // Phase A: out[c,t,w] = sum_v pre_s[c,t,v] * Adyn[c][v,w]
  for (int ci = 0; ci < 8; ++ci) {
    int c = wave * 8 + ci;
    v16bf a = load_a16x32(Ps + c * (16 * VP), VP);
    const bf16* bt = AdynT + ((size_t)(n * KM + k * MID + c)) * (VP * VP);
#pragma unroll
    for (int wt = 0; wt < 2; ++wt) {
      v16bf b = load_b32x16(bt + wt * 16 * VP, VP);
      v8f acc = wmma_bf16(a, b, zero);
#pragma unroll
      for (int r = 0; r < 8; ++r) {
        int tl = r + ((lane < 16) ? 0 : 8);
        int w = wt * 16 + (lane & 15);
        lds[(c * 16 + tl) * VP + w] = acc[r];
      }
    }
  }
  __syncthreads();
  // Phase B: out1[c,t,w] = sum_v pre_s[c,t,v] * A1[t][v,w]  (per-t B)
  float p0 = pai[0];
  int ct = wave >> 1, wt = wave & 1;
  for (int tl = 0; tl < 16; ++tl) {
    int tg = t0 + tl;
    v16bf a = load_a16x32(Ps + (ct * 16) * (16 * VP) + tl * VP, 16 * VP);
    const bf16* bt = A1T + ((size_t)((n * K3 + k) * TT + tg)) * (VP * VP);
    v16bf b = load_b32x16(bt + wt * 16 * VP, VP);
    v8f acc = wmma_bf16(a, b, zero);
#pragma unroll
    for (int r = 0; r < 8; ++r) {
      int cl = ct * 16 + r + ((lane < 16) ? 0 : 8);
      int w = wt * 16 + (lane & 15);
      lds[(cl * 16 + tl) * VP + w] += p0 * acc[r];
    }
  }
  __syncthreads();
  // store to outc[n][tv][km] (bf16, K-major for post GEMM B operand)
  for (int e = threadIdx.x; e < MID * 16 * VV; e += 256) {
    int c = e / (16 * VV);
    int r = e % (16 * VV);
    int tl = r / VV, w = r % VV;
    float val = lds[(c * 16 + tl) * VP + w];
    int j = (t0 + tl) * VV + w;
    outc[((size_t)n * TV + j) * KM + k * MID + c] = (bf16)val;
  }
}

// ---- post GEMM + bn + residual + relu -> d_out, TDM-staged B ----
__global__ void k_post_gemm(const bf16* outc, const bf16* wpost,
                            const float* out_sc, const float* out_sh,
                            const float* x, float* out) {
  __shared__ bf16 Bs[16 * KM];                    // 6KB B tile [col][K]
  int n = blockIdx.y, nt = blockIdx.x;
  int wave = threadIdx.x >> 5, lane = threadIdx.x & 31;
  const bf16* bbase = outc + ((size_t)n * TV + nt * 16) * KM;
  if (threadIdx.x < 32) {
    tdm_load_2d(lds_addr_of(Bs), bbase, (16 * KM * 2) / 8, 1, (16 * KM * 2) / 8);
    __builtin_amdgcn_s_wait_tensorcnt(0);
  }
  __syncthreads();
  v8f zero = {};
  v8f acc[4];
#pragma unroll
  for (int i = 0; i < 4; ++i) acc[i] = zero;
  for (int kt = 0; kt < KM / 32; ++kt) {
    v16bf b = load_b32x16(Bs + kt * 32, KM);
#pragma unroll
    for (int i = 0; i < 4; ++i) {
      int mt = wave + i * 4;                      // 16 m-tiles over 4 waves
      v16bf a = load_a16x32(wpost + mt * 16 * KM + kt * 32, KM);
      acc[i] = wmma_bf16(a, b, acc[i]);
    }
  }
#pragma unroll
  for (int i = 0; i < 4; ++i) {
    int mt = wave + i * 4;
#pragma unroll
    for (int r = 0; r < 8; ++r) {
      int m = mt * 16 + r + ((lane < 16) ? 0 : 8);
      int j = nt * 16 + (lane & 15);
      size_t idx = ((size_t)n * CC + m) * TV + j;
      float val = acc[i][r] * out_sc[m] + out_sh[m] + x[idx];
      out[idx] = fmaxf(val, 0.f);
    }
  }
}

extern "C" void kernel_launch(void* const* d_in, const int* in_sizes, int n_in,
                              void* d_out, int out_size, void* d_ws, size_t ws_size,
                              hipStream_t stream) {
  (void)in_sizes; (void)n_in; (void)out_size; (void)ws_size;
  const float* x      = (const float*)d_in[0];
  const float* A      = (const float*)d_in[1];
  const float* pre_w  = (const float*)d_in[2];
  const float* pre_b  = (const float*)d_in[3];
  const float* bn_pre = (const float*)d_in[4];
  const float* se1_w  = (const float*)d_in[5];
  const float* se1_b  = (const float*)d_in[6];
  const float* se_bn1 = (const float*)d_in[7];
  const float* seh_w  = (const float*)d_in[8];
  const float* seh_b  = (const float*)d_in[9];
  const float* sew_w  = (const float*)d_in[10];
  const float* sew_b  = (const float*)d_in[11];
  const float* c1_w   = (const float*)d_in[12];
  const float* c1_b   = (const float*)d_in[13];
  const float* c2_w   = (const float*)d_in[14];
  const float* c2_b   = (const float*)d_in[15];
  const float* c3_w   = (const float*)d_in[16];
  const float* c3_b   = (const float*)d_in[17];
  const float* post_w = (const float*)d_in[18];
  const float* post_b = (const float*)d_in[19];
  const float* alpha  = (const float*)d_in[20];
  const float* pai    = (const float*)d_in[21];
  const float* bn_out = (const float*)d_in[22];
  float* out = (float*)d_out;

  char* base = (char*)d_ws;
  size_t off = 0;
  auto alloc = [&](size_t bytes) -> char* {
    char* p = base + off;
    off = (off + bytes + 255) & ~(size_t)255;
    return p;
  };
  bf16*  xb     = (bf16*) alloc((size_t)NN * TV * CC * 2);       // 105 MB; reused as outc
  bf16*  outc   = xb;                                            // alias (xb dead after pre GEMM)
  bf16*  pre_s  = (bf16*) alloc((size_t)NN * KM * TT * VP * 2);  // 100 MB
  bf16*  AdynT  = (bf16*) alloc((size_t)NN * KM * VP * VP * 2);  // 25 MB
  bf16*  A1T    = (bf16*) alloc((size_t)NN * K3 * TT * VP * VP * 2); // 50 MB
  float* tmp    = (float*)alloc((size_t)NN * CC * VV * 4);
  float* x3     = (float*)alloc((size_t)NN * K3 * TT * VV * 4);
  float* xh     = (float*)alloc((size_t)NN * KM * TT * 4);
  float* xw     = (float*)alloc((size_t)NN * KM * VV * 4);
  float* a_h    = (float*)alloc((size_t)NN * KM * TT * 4);
  float* a_w    = (float*)alloc((size_t)NN * KM * VV * 4);
  bf16*  wpre   = (bf16*) alloc((size_t)KM * CC * 2);
  bf16*  wpost  = (bf16*) alloc((size_t)CC * KM * 2);
  float* pre_sc = (float*)alloc(KM * 4);
  float* pre_sh = (float*)alloc(KM * 4);
  float* out_sc = (float*)alloc(CC * 4);
  float* out_sh = (float*)alloc(CC * 4);
  float* se1_sc = (float*)alloc(MIP * 4);
  float* se1_sh = (float*)alloc(MIP * 4);
  float* w3r    = (float*)alloc(K3 * CC * 4);
  float* b3r    = (float*)alloc(K3 * 4);

  k_prep<<<1, 256, 0, stream>>>(pre_w, pre_b, bn_pre, se1_b, se_bn1, c3_w, c3_b,
                                post_w, post_b, bn_out,
                                pre_sc, pre_sh, out_sc, out_sh, se1_sc, se1_sh,
                                w3r, b3r, wpre, wpost);
  k_cvt<<<dim3((TV + 255) / 256, NN), 256, 0, stream>>>(x, xb);
  k_gate_tmp<<<NN, 256, 0, stream>>>(x, tmp);
  k_x3<<<dim3(TT, NN), 128, 0, stream>>>(x, w3r, b3r, x3);
  k_adyn<<<NN, 256, 0, stream>>>(tmp, c1_w, c1_b, c2_w, c2_b, A, alpha, AdynT);
  k_a1<<<dim3(TT, NN), 256, 0, stream>>>(x3, A1T);
  k_pre_gemm<<<dim3(TV / 16, NN), 128, 0, stream>>>(xb, wpre, pre_sc, pre_sh, pre_s);
  k_xhxw<<<dim3(KM, NN), 128, 0, stream>>>(pre_s, xh, xw);
  k_se<<<NN, 256, 0, stream>>>(xh, xw, se1_w, se1_sc, se1_sh,
                               seh_w, seh_b, sew_w, sew_b, a_h, a_w);
  k_scale<<<(NN * KM * TT) / 256, 256, 0, stream>>>(pre_s, a_h, a_w);
  k_einsum<<<dim3(TT / 16, K3, NN), 256,
             64 * 16 * VP * sizeof(float) + MID * 16 * VP * sizeof(bf16), stream>>>(
      pre_s, AdynT, A1T, pai, outc);
  k_post_gemm<<<dim3(TV / 16, NN), 128, 0, stream>>>(outc, wpost, out_sc, out_sh, x, out);
}
